// SoftSymmetricAlignment_31868657336518
// MI455X (gfx1250) — matrix operand order repaired and checked by reading, not compiled, not run on hardware
//
#include <hip/hip_runtime.h>
#include <hip/hip_bf16.h>

// ---- problem constants (B=8, L=1024, D_in=512, D_emb=128) ----
#define BATCH 8
#define LSEQ 1024
#define DIN 512
#define DEMB 128
#define ROWS (BATCH * 2 * LSEQ)   // 16384 rows of the projection GEMM
#define NEGINF (-1000000000.0f)

typedef __attribute__((ext_vector_type(16))) __bf16 v16bf;
typedef __attribute__((ext_vector_type(8)))  __bf16 v8bf;
typedef __attribute__((ext_vector_type(8)))  float  v8f;

static __device__ __forceinline__ __bf16 f2bf(float f) {
  union { float f; unsigned u; } v; v.f = f;
  unsigned r = v.u + 0x7FFFu + ((v.u >> 16) & 1u);   // round-to-nearest-even
  unsigned short h = (unsigned short)(r >> 16);
  return __builtin_bit_cast(__bf16, h);
}

// ---- CDNA5 async global->LDS copy (ASYNCcnt-tracked), with safe fallback ----
#if __has_builtin(__builtin_amdgcn_global_load_async_to_lds_b128)
#define HAVE_ASYNC_CP 1
#endif

#ifdef HAVE_ASYNC_CP
// Diagnostic from round 2 shows the builtin takes pointers to
// 'int __attribute__((vector_size(16)))' in address spaces 1 (global) / 3 (LDS).
typedef int v4i_t __attribute__((vector_size(16)));
typedef v4i_t __attribute__((address_space(1))) gv4i;
typedef v4i_t __attribute__((address_space(3))) lv4i;
#endif

static __device__ __forceinline__ void cp_b128(const __bf16* __restrict__ g,
                                               __bf16* l) {
#ifdef HAVE_ASYNC_CP
  __builtin_amdgcn_global_load_async_to_lds_b128((gv4i*)g, (lv4i*)l, 0, 0);
#else
  *(v8bf*)l = *(const v8bf*)g;   // load + ds_store fallback
#endif
}

static __device__ __forceinline__ void async_wait() {
#ifdef HAVE_ASYNC_CP
#if __has_builtin(__builtin_amdgcn_s_wait_asynccnt)
  __builtin_amdgcn_s_wait_asynccnt(0);
#else
  asm volatile("s_wait_asynccnt 0" ::: "memory");
#endif
#endif
}

// K0: f32 -> bf16 copies of embeddings, plus bf16 transpose of W (so WMMA
// B-fragments are 16B loads contiguous in k).
__global__ __launch_bounds__(256) void k_convert(const float* __restrict__ E,
                                                 const float* __restrict__ W,
                                                 __bf16* __restrict__ ebf,
                                                 __bf16* __restrict__ Wt) {
  size_t i = (size_t)blockIdx.x * 256 + threadIdx.x;
  const size_t NE = (size_t)ROWS * DIN;
  if (i < NE) {
    ebf[i] = f2bf(E[i]);
  } else {
    size_t j = i - NE;
    if (j < (size_t)DIN * DEMB) {
      int k = (int)(j >> 7);        // W is [DIN][DEMB] row-major
      int n = (int)(j & 127);
      Wt[(size_t)n * DIN + k] = f2bf(W[j]);   // Wt is [DEMB][DIN]
    }
  }
}

// K1: projection GEMM  emb[16384x128] = ebf[16384x512] @ W + bias
// Block: 8 waves, 128 rows. B k-slab (32x128) double-buffered in LDS and
// shared by all 8 waves; async staged. Also emits per-row |emb|^2 (nsq).
#define PJ_STRIDE 40   // padded LDS row stride (elements) to avoid bank conflicts
__global__ __launch_bounds__(256) void k_proj(const __bf16* __restrict__ ebf,
                                              const __bf16* __restrict__ Wt,
                                              const float* __restrict__ bias,
                                              __bf16* __restrict__ embbf,
                                              float* __restrict__ nsq) {
  __shared__ __bf16 sB[2][DEMB * PJ_STRIDE];   // 2 x 10 KB
  const int tid = threadIdx.x;
  const int lane = tid & 31;
  const int wid = tid >> 5;
  const int row0 = (blockIdx.x * 8 + wid) << 4;
  const int m = lane & 15, h = lane >> 4;

  float bvals[8];
#pragma unroll
  for (int nt = 0; nt < 8; ++nt) bvals[nt] = bias[nt * 16 + m];

  v8f acc[8];
#pragma unroll
  for (int t = 0; t < 8; ++t) acc[t] = v8f{};

  // cooperative stage of k-slab kt (128 rows x 32 bf16 = 512 16B chunks)
  auto stage = [&](int kt, int buf) {
#pragma unroll
    for (int c = tid; c < 512; c += 256) {
      int n = c >> 2, part = c & 3;
      cp_b128(Wt + (size_t)n * DIN + kt * 32 + part * 8,
              &sB[buf][n * PJ_STRIDE + part * 8]);
    }
  };

  stage(0, 0);
  const __bf16* arow = ebf + (size_t)(row0 + m) * DIN;
  for (int kt = 0; kt < 16; ++kt) {
    async_wait();
    __syncthreads();
    if (kt + 1 < 16) stage(kt + 1, (kt + 1) & 1);
    const __bf16* bs = sB[kt & 1];
    const int k0 = kt * 32;
    union { v16bf v; v8bf h8[2]; } ua;
    ua.h8[0] = *(const v8bf*)(arow + k0 + h * 8);
    ua.h8[1] = *(const v8bf*)(arow + k0 + 16 + h * 8);
#pragma unroll
    for (int nt = 0; nt < 8; ++nt) {
      union { v16bf v; v8bf h8[2]; } ub;
      const __bf16* bp = bs + (nt * 16 + m) * PJ_STRIDE + h * 8;
      ub.h8[0] = *(const v8bf*)(bp);
      ub.h8[1] = *(const v8bf*)(bp + 16);
      acc[nt] = __builtin_amdgcn_wmma_f32_16x16x32_bf16(
          false, ua.v, false, ub.v, (short)0, acc[nt], false, false);
    }
  }

  // store bf16 emb + accumulate row norms
  float rowsq[8];
#pragma unroll
  for (int i = 0; i < 8; ++i) rowsq[i] = 0.0f;
#pragma unroll
  for (int nt = 0; nt < 8; ++nt) {
#pragma unroll
    for (int i = 0; i < 8; ++i) {
      int r = row0 + i + h * 8;
      int c = nt * 16 + m;
      float v = acc[nt][i] + bvals[nt];
      embbf[(size_t)r * DEMB + c] = f2bf(v);
      rowsq[i] += v * v;
    }
  }
  // reduce over the 16 column-lanes of each h-group
#pragma unroll
  for (int i = 0; i < 8; ++i) {
    float s = rowsq[i];
    s += __shfl_xor(s, 1, 32);
    s += __shfl_xor(s, 2, 32);
    s += __shfl_xor(s, 4, 32);
    s += __shfl_xor(s, 8, 32);
    if (m == 0) nsq[row0 + i + h * 8] = s;
  }
}

// K2: pair_dist[b][l][m] = |x_l|^2 + |y_m|^2 - 2 <x_l, y_m>   (WMMA over D=128)
// Block computes a 128x128 output tile; y-tile (128x128 bf16) staged in LDS
// once and reused by all 8 waves x 8 n-tiles.
#define PR_STRIDE 136  // padded LDS row stride (elements)
__global__ __launch_bounds__(256) void k_pair(const __bf16* __restrict__ embbf,
                                              const float* __restrict__ nsq,
                                              float* __restrict__ pdist) {
  __shared__ __bf16 sY[DEMB * PR_STRIDE];   // 34 KB
  const int tid = threadIdx.x;
  const int lane = tid & 31;
  const int wid = tid >> 5;
  const int b = blockIdx.x >> 6;
  const int rem = blockIdx.x & 63;
  const int l0 = ((rem >> 3) << 7) + (wid << 4);   // this wave's 16 rows
  const int m0 = (rem & 7) << 7;                   // block's 128 columns
  const int m = lane & 15, h = lane >> 4;

  const __bf16* x = embbf + (size_t)b * 2 * LSEQ * DEMB;
  const __bf16* y = x + (size_t)LSEQ * DEMB;

  // stage y[m0..m0+127][0..127] : 2048 16B chunks, 8 per thread
#pragma unroll
  for (int c = tid; c < 2048; c += 256) {
    int n = c >> 4, part = c & 15;
    cp_b128(y + (size_t)(m0 + n) * DEMB + part * 8,
            &sY[n * PR_STRIDE + part * 8]);
  }
  async_wait();
  __syncthreads();

  v8f acc[8];
#pragma unroll
  for (int t = 0; t < 8; ++t) acc[t] = v8f{};

  const __bf16* arow = x + (size_t)(l0 + m) * DEMB;
#pragma unroll
  for (int k0 = 0; k0 < DEMB; k0 += 32) {
    union { v16bf v; v8bf h8[2]; } ua;
    ua.h8[0] = *(const v8bf*)(arow + k0 + h * 8);
    ua.h8[1] = *(const v8bf*)(arow + k0 + 16 + h * 8);
#pragma unroll
    for (int nt = 0; nt < 8; ++nt) {
      union { v16bf v; v8bf h8[2]; } ub;
      const __bf16* bp = sY + (nt * 16 + m) * PR_STRIDE + k0 + h * 8;
      ub.h8[0] = *(const v8bf*)(bp);
      ub.h8[1] = *(const v8bf*)(bp + 16);
      acc[nt] = __builtin_amdgcn_wmma_f32_16x16x32_bf16(
          false, ua.v, false, ub.v, (short)0, acc[nt], false, false);
    }
  }

  const float* nx = nsq + (size_t)b * 2 * LSEQ;
  const float* ny = nx + LSEQ;
  float* pd = pdist + ((size_t)b << 20);
#pragma unroll
  for (int nt = 0; nt < 8; ++nt) {
#pragma unroll
    for (int i = 0; i < 8; ++i) {
      int l = l0 + i + h * 8;
      int mm = m0 + nt * 16 + m;
      pd[(size_t)l * LSEQ + mm] = nx[l] + ny[mm] - 2.0f * acc[nt][i];
    }
  }
}

// K3: per-row softmax stats (axis=2): max + sum of exp (block per row)
__global__ __launch_bounds__(256) void k_rowstats(const float* __restrict__ pdist,
                                                  const unsigned char* __restrict__ mask,
                                                  float* __restrict__ rmax,
                                                  float* __restrict__ rsum) {
  __shared__ float sm[256];
  const int row = blockIdx.x;          // 0..8191
  const int b = row >> 10, l = row & 1023;
  const float* pd = pdist + ((size_t)b << 20) + (size_t)l * LSEQ;
  const unsigned char* mx = mask + (size_t)b * 2 * LSEQ;
  const unsigned char* my = mx + LSEQ;
  const bool ml = mx[l] != 0;

  float lm = NEGINF;
  for (int c = threadIdx.x; c < LSEQ; c += 256) {
    float logit = (ml && my[c]) ? -pd[c] : NEGINF;
    lm = fmaxf(lm, logit);
  }
  sm[threadIdx.x] = lm; __syncthreads();
  for (int s = 128; s > 0; s >>= 1) {
    if ((int)threadIdx.x < s) sm[threadIdx.x] = fmaxf(sm[threadIdx.x], sm[threadIdx.x + s]);
    __syncthreads();
  }
  const float M = sm[0];
  __syncthreads();

  float ls = 0.0f;
  for (int c = threadIdx.x; c < LSEQ; c += 256) {
    float logit = (ml && my[c]) ? -pd[c] : NEGINF;
    ls += __expf(logit - M);
  }
  sm[threadIdx.x] = ls; __syncthreads();
  for (int s = 128; s > 0; s >>= 1) {
    if ((int)threadIdx.x < s) sm[threadIdx.x] += sm[threadIdx.x + s];
    __syncthreads();
  }
  if (threadIdx.x == 0) { rmax[row] = M; rsum[row] = sm[0]; }
}

// K4: per-column softmax stats (axis=1): online max/sum, thread per column
__global__ __launch_bounds__(256) void k_colstats(const float* __restrict__ pdist,
                                                  const unsigned char* __restrict__ mask,
                                                  float* __restrict__ cmax,
                                                  float* __restrict__ csum) {
  const int col = (int)(blockIdx.x * blockDim.x + threadIdx.x);  // 0..8191
  const int b = col >> 10, c = col & 1023;
  const float* pd = pdist + ((size_t)b << 20) + c;
  const unsigned char* mx = mask + (size_t)b * 2 * LSEQ;
  const bool mc = mx[LSEQ + c] != 0;

  float M = -INFINITY, S = 0.0f;
  for (int l = 0; l < LSEQ; ++l) {
    float logit = (mc && mx[l]) ? -pd[(size_t)l * LSEQ] : NEGINF;
    float nM = fmaxf(M, logit);
    S = S * __expf(M - nM) + __expf(logit - nM);
    M = nM;
  }
  cmax[col] = M; csum[col] = S;
}

// K5: per-row partial of numerator/denominator of the score
__global__ __launch_bounds__(256) void k_accum(const float* __restrict__ pdist,
                                               const unsigned char* __restrict__ mask,
                                               const float* __restrict__ rmax,
                                               const float* __restrict__ rsum,
                                               const float* __restrict__ cmax,
                                               const float* __restrict__ csum,
                                               float* __restrict__ rnum,
                                               float* __restrict__ rden) {
  __shared__ float sn[256], sd[256];
  const int row = blockIdx.x;
  const int b = row >> 10, l = row & 1023;
  const float* pd = pdist + ((size_t)b << 20) + (size_t)l * LSEQ;
  const unsigned char* mx = mask + (size_t)b * 2 * LSEQ;
  const unsigned char* my = mx + LSEQ;
  const bool ml = mx[l] != 0;
  const float rM = rmax[row], rS = rsum[row];

  float ln = 0.0f, ld = 0.0f;
  for (int c = threadIdx.x; c < LSEQ; c += 256) {
    float d = pd[c];
    bool pm = ml && my[c];
    float logit = pm ? -d : NEGINF;
    float a = __expf(logit - rM) / rS;
    float bb = __expf(logit - cmax[(b << 10) + c]) / csum[(b << 10) + c];
    float att = pm ? (a + bb - a * bb) : 0.0f;
    ln += att * d;
    ld += att;
  }
  sn[threadIdx.x] = ln; sd[threadIdx.x] = ld; __syncthreads();
  for (int s = 128; s > 0; s >>= 1) {
    if ((int)threadIdx.x < s) {
      sn[threadIdx.x] += sn[threadIdx.x + s];
      sd[threadIdx.x] += sd[threadIdx.x + s];
    }
    __syncthreads();
  }
  if (threadIdx.x == 0) { rnum[row] = sn[0]; rden[row] = sd[0]; }
}

// K6: deterministic per-batch finalize: scores[b] = -num/den
__global__ __launch_bounds__(256) void k_final(const float* __restrict__ rnum,
                                               const float* __restrict__ rden,
                                               float* __restrict__ out) {
  __shared__ float sn[256], sd[256];
  const int b = blockIdx.x;
  float n = 0.0f, d = 0.0f;
  for (int l = threadIdx.x; l < LSEQ; l += 256) {
    n += rnum[b * LSEQ + l];
    d += rden[b * LSEQ + l];
  }
  sn[threadIdx.x] = n; sd[threadIdx.x] = d; __syncthreads();
  for (int s = 128; s > 0; s >>= 1) {
    if ((int)threadIdx.x < s) {
      sn[threadIdx.x] += sn[threadIdx.x + s];
      sd[threadIdx.x] += sd[threadIdx.x + s];
    }
    __syncthreads();
  }
  if (threadIdx.x == 0) out[b] = -sn[0] / sd[0];
}

extern "C" void kernel_launch(void* const* d_in, const int* in_sizes, int n_in,
                              void* d_out, int out_size, void* d_ws, size_t ws_size,
                              hipStream_t stream) {
  const float* E = (const float*)d_in[0];                    // (8,2,1024,512) f32
  const unsigned char* mask = (const unsigned char*)d_in[1]; // (8,2,1024) bool
  const float* W = (const float*)d_in[2];                    // (512,128) f32
  const float* bias = (const float*)d_in[3];                 // (128,) f32
  float* out = (float*)d_out;                                // (8,) f32

  // ---- workspace carve-up (256B aligned) ----
  char* base = (char*)d_ws;
  size_t off = 0;
  auto carve = [&](size_t bytes) -> char* {
    char* p = base + off;
    off = (off + bytes + 255) & ~(size_t)255;
    return p;
  };
  __bf16* ebf   = (__bf16*)carve((size_t)ROWS * DIN * 2);          // 16 MB
  __bf16* Wt    = (__bf16*)carve((size_t)DEMB * DIN * 2);          // 128 KB
  __bf16* embbf = (__bf16*)carve((size_t)ROWS * DEMB * 2);         // 4 MB
  float*  nsq   = (float*) carve((size_t)ROWS * 4);                // 64 KB
  float*  pdist = (float*) carve((size_t)BATCH * LSEQ * LSEQ * 4); // 32 MB
  float*  rmax  = (float*) carve((size_t)BATCH * LSEQ * 4);
  float*  rsum  = (float*) carve((size_t)BATCH * LSEQ * 4);
  float*  cmax  = (float*) carve((size_t)BATCH * LSEQ * 4);
  float*  csum  = (float*) carve((size_t)BATCH * LSEQ * 4);
  float*  rnum  = (float*) carve((size_t)BATCH * LSEQ * 4);
  float*  rden  = (float*) carve((size_t)BATCH * LSEQ * 4);
  (void)ws_size; (void)in_sizes; (void)n_in; (void)out_size;

  // K0: convert inputs to bf16 (+ transposed W)
  {
    size_t total = (size_t)ROWS * DIN + (size_t)DIN * DEMB;
    k_convert<<<dim3((unsigned)((total + 255) / 256)), 256, 0, stream>>>(E, W, ebf, Wt);
  }
  // K1: projection GEMM (WMMA bf16, LDS double-buffered B): 128 blocks x 8 waves
  k_proj<<<dim3(ROWS / 128), 256, 0, stream>>>(ebf, Wt, bias, embbf, nsq);
  // K2: pair distances (WMMA bf16, LDS y-tile): 8 batches x 64 tiles
  k_pair<<<dim3(BATCH * 64), 256, 0, stream>>>(embbf, nsq, pdist);
  // K3: row softmax stats (block per row)
  k_rowstats<<<dim3(BATCH * LSEQ), 256, 0, stream>>>(pdist, mask, rmax, rsum);
  // K4: column softmax stats (thread per column)
  k_colstats<<<dim3(BATCH * LSEQ / 256), 256, 0, stream>>>(pdist, mask, cmax, csum);
  // K5: per-row score partials
  k_accum<<<dim3(BATCH * LSEQ), 256, 0, stream>>>(pdist, mask, rmax, rsum, cmax, csum, rnum, rden);
  // K6: per-batch finalize
  k_final<<<dim3(BATCH), 256, 0, stream>>>(rnum, rden, out);
}
